// SambaReadout_78357383348651
// MI455X (gfx1250) — matrix operand order repaired and checked by reading, not compiled, not run on hardware
//
#include <hip/hip_runtime.h>
#include <hip/hip_bf16.h>

// ---------------------------------------------------------------------------
// SambaReadout for MI455X (gfx1250, wave32, WMMA).
// - bf16 WMMA (v_wmma_f32_16x16x32_bf16, f32 accum) for all five GEMMs:
//   memory roofline (~425MB @ 23.3TB/s ~ 18us) needs >2.3 PFLOPS matrix math.
// - Weights cast f32->bf16 ONCE and stored TRANSPOSED (N x K): GEMM staging
//   is pure 16B traffic.
// - GEMM staging uses CDNA5 async global->LDS copies
//   (global_load_async_to_lds_b128, ASYNCcnt): no VGPR round-trip, copy of
//   tile k+32 hidden behind the WMMAs of tile k; one barrier per K-step;
//   K-loop unrolled x2 so double-buffer addresses are immediates.
// - 128x128 block tile, wave grid 4x2: each wave = 32x64 strip -> 8 WMMAs per
//   12 ds_load_b128 (2 A-frags x 4 B-frags), halving LDS+barrier cost/FLOP.
// - Linear interpolation commutes with the affine W2 map: interpolate the 1MB
//   `mid` tensor, not the 65MB `out` tensor; final GEMM writes logits (262MB)
//   straight to d_out.
// ---------------------------------------------------------------------------

typedef __attribute__((ext_vector_type(16))) __bf16   v16bf;
typedef __attribute__((ext_vector_type(8)))  float    v8f;
typedef __attribute__((ext_vector_type(4)))  unsigned u32x4;   // trivially-copyable 16B

#define L_      6
#define B_      2
#define S_      1024
#define D_      8192
#define H_      512
#define V_      32000
#define STRIDE_ 4
#define T_      (S_ / STRIDE_)   // 256
#define KQ_     128
#define MID_    (H_ / 2)         // 256

struct bf4 { __bf16 v[4]; };

__device__ inline __bf16 f2bf(float x) { return (__bf16)x; }

// LDS byte offset of a __shared__ object (AS3 pointers are LDS offsets).
typedef __attribute__((address_space(3))) void lds_void;
__device__ inline unsigned lds_addr(const void* p) {
    return (unsigned)(unsigned long long)(lds_void*)p;
}

// One 16B async copy global -> LDS (per lane), tracked by ASYNCcnt.
__device__ inline void async_cp16(unsigned lds_off, const void* gaddr) {
    asm volatile("global_load_async_to_lds_b128 %0, %1, off"
                 :: "v"(lds_off), "v"(gaddr) : "memory");
}
__device__ inline void wait_async0() {
    asm volatile("s_wait_asynccnt 0x0" ::: "memory");
}

// ------------- f32 (KxN) -> bf16 transposed (NxK), LDS-tiled ---------------
__global__ __launch_bounds__(256) void cast_transpose_kernel(
    const float* __restrict__ in, __bf16* __restrict__ out, int K, int N)
{
    __shared__ __bf16 tile[32][33];
    const int n0 = blockIdx.x * 32;
    const int k0 = blockIdx.y * 32;
    const int tx = threadIdx.x & 31;    // fast: n on read, k on write
    const int ty = threadIdx.x >> 5;    // 0..7
#pragma unroll
    for (int i = 0; i < 32; i += 8)     // read coalesced along N
        tile[tx][ty + i] = f2bf(in[(long)(k0 + ty + i) * N + n0 + tx]);
    __syncthreads();
#pragma unroll
    for (int i = 0; i < 32; i += 8)     // write coalesced along K
        out[(long)(n0 + ty + i) * K + k0 + tx] = tile[ty + i][tx];
}

// ----------- stride-sample h, cast to bf16, and fold in layer mean ----------
// out layout: (L_+1, B_*T_, D_); slot L_ holds the mean over layers.
__global__ __launch_bounds__(256) void pack_h_kernel(
    const float* __restrict__ hs, __bf16* __restrict__ hB)
{
    const int groups = (B_ * T_ * D_) / 4;
    int i = blockIdx.x * 256 + threadIdx.x;
    if (i >= groups) return;
    int idx = i * 4;
    int d  = idx % D_;
    int bt = idx / D_;
    int t  = bt % T_;
    int b  = bt / T_;
    long src = ((long)b * S_ + (long)t * STRIDE_) * D_ + d;
    float sx = 0.f, sy = 0.f, sz = 0.f, sw = 0.f;
#pragma unroll
    for (int l = 0; l < L_; ++l) {
        float4 x = *(const float4*)(hs + (long)l * (B_ * S_ * (long)D_) + src);
        bf4 y;
        y.v[0] = f2bf(x.x); y.v[1] = f2bf(x.y); y.v[2] = f2bf(x.z); y.v[3] = f2bf(x.w);
        ((bf4*)(hB + (long)l * (B_ * T_ * (long)D_)))[i] = y;
        sx += x.x; sy += x.y; sz += x.z; sw += x.w;
    }
    const float inv = 1.0f / (float)L_;
    bf4 m;
    m.v[0] = f2bf(sx * inv); m.v[1] = f2bf(sy * inv);
    m.v[2] = f2bf(sz * inv); m.v[3] = f2bf(sw * inv);
    ((bf4*)(hB + (long)L_ * (B_ * T_ * (long)D_)))[i] = m;
}

// ----------------------------- WMMA GEMM -----------------------------------
// C(MxN) = act(A(MxK, bf16 rowmajor) @ B + bias), B given TRANSPOSED
// Bt(NxK, bf16 rowmajor). 256 thr = 8 waves; block tile 128x128, BK=32;
// LDS double-buffered via async global->LDS copies, K-loop unrolled x2.
// Wave (wr,wc) in a 4x2 grid owns a 32x64 strip: 2 A-frags x 4 B-frags ->
// 8 chained v_wmma per K-step per wave.
// A-frag: lane(m,half) holds K = half*16+[0..15] (same order as B-frag)
// -> two contiguous b128 LDS reads.
// B-frag: lane(n,khalf) holds column n, K = khalf*16..+15 -> Bt tile is
// K-contiguous -> two contiguous b128 reads.
// Requires M%128==0, N%128==0, K%64==0 (true for every call site).
template <int RELU, int OUT_BF16>
__global__ __launch_bounds__(256) void gemm_bf16_kernel(
    const __bf16* __restrict__ A, const __bf16* __restrict__ Bt,
    const float* __restrict__ bias, void* __restrict__ Out,
    int M, int N, int K)
{
    __shared__ __attribute__((aligned(16))) __bf16 As[2][128][48];
    __shared__ __attribute__((aligned(16))) __bf16 Bs[2][128][48];

    const int tid  = threadIdx.x;
    const int wave = tid >> 5;
    const int lane = tid & 31;
    const int wr   = wave >> 1;      // 0..3 : 32-row group
    const int wc   = wave & 1;       // 0..1 : 64-col group
    const int half = lane >> 4;      // 0/1
    const int mr   = lane & 15;
    const int m0   = blockIdx.x * 128;
    const int n0   = blockIdx.y * 128;

    // global->LDS staging map: each thread owns 32B of one A row + one B row
    const int srow = tid >> 1;               // 0..127
    const int scol = (tid & 1) * 16;         // 0 or 16
    const __bf16* aG = A  + (long)(m0 + srow) * K + scol;
    const __bf16* bG = Bt + (long)(n0 + srow) * K + scol;
    const unsigned lA0 = lds_addr(&As[0][srow][scol]);
    const unsigned lA1 = lds_addr(&As[1][srow][scol]);
    const unsigned lB0 = lds_addr(&Bs[0][srow][scol]);
    const unsigned lB1 = lds_addr(&Bs[1][srow][scol]);

    v8f acc[8] = {};

    auto stage = [&](unsigned lA, unsigned lB, int k) {
        async_cp16(lA,      aG + k);
        async_cp16(lA + 16, aG + k + 8);
        async_cp16(lB,      bG + k);
        async_cp16(lB + 16, bG + k + 8);
    };
    auto compute = [&](int buf) {
        union F { v16bf v; u32x4 q[2]; };
        F a0, a1, b0, b1, b2, b3;
        const __bf16* ap0 = &As[buf][wr * 32 + mr][0];
        const __bf16* ap1 = &As[buf][wr * 32 + 16 + mr][0];
        a0.q[0] = *(const u32x4*)(ap0 + half * 16);
        a0.q[1] = *(const u32x4*)(ap0 + half * 16 + 8);
        a1.q[0] = *(const u32x4*)(ap1 + half * 16);
        a1.q[1] = *(const u32x4*)(ap1 + half * 16 + 8);
        const __bf16* bp = &Bs[buf][wc * 64 + mr][half * 16];
        b0.q[0] = *(const u32x4*)(bp);
        b0.q[1] = *(const u32x4*)(bp + 8);
        b1.q[0] = *(const u32x4*)(bp + 16 * 48);
        b1.q[1] = *(const u32x4*)(bp + 16 * 48 + 8);
        b2.q[0] = *(const u32x4*)(bp + 32 * 48);
        b2.q[1] = *(const u32x4*)(bp + 32 * 48 + 8);
        b3.q[0] = *(const u32x4*)(bp + 48 * 48);
        b3.q[1] = *(const u32x4*)(bp + 48 * 48 + 8);
        acc[0] = __builtin_amdgcn_wmma_f32_16x16x32_bf16(
            false, a0.v, false, b0.v, (short)0, acc[0], false, false);
        acc[1] = __builtin_amdgcn_wmma_f32_16x16x32_bf16(
            false, a0.v, false, b1.v, (short)0, acc[1], false, false);
        acc[2] = __builtin_amdgcn_wmma_f32_16x16x32_bf16(
            false, a0.v, false, b2.v, (short)0, acc[2], false, false);
        acc[3] = __builtin_amdgcn_wmma_f32_16x16x32_bf16(
            false, a0.v, false, b3.v, (short)0, acc[3], false, false);
        acc[4] = __builtin_amdgcn_wmma_f32_16x16x32_bf16(
            false, a1.v, false, b0.v, (short)0, acc[4], false, false);
        acc[5] = __builtin_amdgcn_wmma_f32_16x16x32_bf16(
            false, a1.v, false, b1.v, (short)0, acc[5], false, false);
        acc[6] = __builtin_amdgcn_wmma_f32_16x16x32_bf16(
            false, a1.v, false, b2.v, (short)0, acc[6], false, false);
        acc[7] = __builtin_amdgcn_wmma_f32_16x16x32_bf16(
            false, a1.v, false, b3.v, (short)0, acc[7], false, false);
    };

    // prologue: stage K-tile 0 into buffer 0
    stage(lA0, lB0, 0);
    wait_async0();
    __syncthreads();

    for (int k0 = 0; k0 < K; k0 += 64) {
        // copy tile (k0+32) -> buf1 while computing tile k0 from buf0
        stage(lA1, lB1, k0 + 32);
        // warm GL2 two tiles ahead (speculative; OOB prefetch is dropped)
        __builtin_prefetch(aG + k0 + 96, 0, 0);
        __builtin_prefetch(bG + k0 + 96, 0, 0);
        compute(0);
        wait_async0();
        __syncthreads();

        // copy tile (k0+64) -> buf0 while computing tile (k0+32) from buf1
        if (k0 + 64 < K) stage(lA0, lB0, k0 + 64);
        compute(1);
        wait_async0();
        __syncthreads();
    }

    // Epilogue: bias + optional ReLU + optional bf16 downconvert.
    // C/D layout: n = lane%16, m = vgpr + 8*(lane>=16).
#pragma unroll
    for (int i = 0; i < 2; ++i) {
#pragma unroll
        for (int j = 0; j < 4; ++j) {
            int col = n0 + wc * 64 + j * 16 + mr;
            float bv = bias[col];
#pragma unroll
            for (int r = 0; r < 8; ++r) {
                int row = m0 + wr * 32 + i * 16 + r + 8 * half;
                float val = acc[i * 4 + j][r] + bv;
                if (RELU) val = fmaxf(val, 0.0f);
                if (OUT_BF16) ((__bf16*)Out)[(long)row * N + col] = f2bf(val);
                else          ((float*)Out)[(long)row * N + col]  = val;
            }
        }
    }
}

// -------- per-(b,t): scores -> softmax over L -> weighted agg of v ----------
__global__ __launch_bounds__(128) void attn_kernel(
    const float* __restrict__ q, const float* __restrict__ k,
    const float* __restrict__ v, __bf16* __restrict__ aggB)
{
    const int bt   = blockIdx.x;        // 0..B*T-1
    const int tid  = threadIdx.x;       // 0..127
    const int wave = tid >> 5, lane = tid & 31;
    __shared__ float red[L_][4];
    __shared__ float sw[L_];

    float qv = q[bt * KQ_ + tid];
#pragma unroll
    for (int l = 0; l < L_; ++l) {
        float p = qv * k[((long)l * (B_ * T_) + bt) * KQ_ + tid];
#pragma unroll
        for (int off = 16; off > 0; off >>= 1) p += __shfl_down(p, off, 32);
        if (lane == 0) red[l][wave] = p;
    }
    __syncthreads();
    if (tid == 0) {
        const float scale = 0.08838834764831845f;   // 128^-0.5
        float s[L_], mx = -1e30f;
#pragma unroll
        for (int l = 0; l < L_; ++l) {
            s[l] = (red[l][0] + red[l][1] + red[l][2] + red[l][3]) * scale;
            mx = fmaxf(mx, s[l]);
        }
        float sum = 0.f;
#pragma unroll
        for (int l = 0; l < L_; ++l) { s[l] = __expf(s[l] - mx); sum += s[l]; }
        float inv = 1.0f / sum;
#pragma unroll
        for (int l = 0; l < L_; ++l) sw[l] = s[l] * inv;
    }
    __syncthreads();
    for (int h = tid; h < H_; h += 128) {
        float a = 0.f;
#pragma unroll
        for (int l = 0; l < L_; ++l)
            a += sw[l] * v[((long)l * (B_ * T_) + bt) * H_ + h];
        aggB[(long)bt * H_ + h] = f2bf(a);
    }
}

// ------ interpolate mid (B,T,MID) -> midI (B,S,MID); commutes with @W2 ------
__global__ __launch_bounds__(256) void interp_kernel(
    const __bf16* __restrict__ mid, __bf16* __restrict__ midI)
{
    int bs = blockIdx.x;                 // 0..B*S-1
    int s = bs % S_, b = bs / S_;
    float src = (s + 0.5f) * ((float)T_ / (float)S_) - 0.5f;
    src = fminf(fmaxf(src, 0.0f), (float)(T_ - 1));
    int lo = (int)floorf(src);
    int hi = min(lo + 1, T_ - 1);
    float f = src - (float)lo;
    int c = threadIdx.x;                 // 0..MID_-1
    float vlo = (float)mid[((long)b * T_ + lo) * MID_ + c];
    float vhi = (float)mid[((long)b * T_ + hi) * MID_ + c];
    midI[(long)bs * MID_ + c] = f2bf((1.0f - f) * vlo + f * vhi);
}

// ------------- tuple tail: sampled_indices = arange(0,S,stride) -------------
__global__ void tail_kernel(float* __restrict__ out, int out_size)
{
    int i = threadIdx.x;
    long base = (long)B_ * S_ * V_;
    if (i < T_ && base + i < (long)out_size) out[base + i] = (float)(i * STRIDE_);
}

// ---------------------------------------------------------------------------
extern "C" void kernel_launch(void* const* d_in, const int* in_sizes, int n_in,
                              void* d_out, int out_size, void* d_ws, size_t ws_size,
                              hipStream_t stream)
{
    const float* hs = (const float*)d_in[0];
    const float* Wq = (const float*)d_in[1];
    const float* bq = (const float*)d_in[2];
    const float* Wk = (const float*)d_in[3];
    const float* bk = (const float*)d_in[4];
    const float* Wv = (const float*)d_in[5];
    const float* bv = (const float*)d_in[6];
    const float* W1 = (const float*)d_in[7];
    const float* b1 = (const float*)d_in[8];
    const float* W2 = (const float*)d_in[9];
    const float* b2 = (const float*)d_in[10];
    float* out = (float*)d_out;
    (void)in_sizes; (void)n_in; (void)ws_size;

    // workspace carve-up (~97 MB total)
    char* ws = (char*)d_ws;
    size_t off = 0;
    auto take = [&](size_t bytes) -> char* {
        char* p = ws + off;
        off = (off + bytes + 255) & ~(size_t)255;
        return p;
    };
    __bf16* hB   = (__bf16*)take((size_t)(L_ + 1) * B_ * T_ * D_ * 2);  // 58.7MB
    __bf16* WqT  = (__bf16*)take((size_t)D_ * KQ_ * 2);
    __bf16* WkT  = (__bf16*)take((size_t)D_ * KQ_ * 2);
    __bf16* WvT  = (__bf16*)take((size_t)D_ * H_ * 2);
    __bf16* W1T  = (__bf16*)take((size_t)H_ * MID_ * 2);
    __bf16* W2T  = (__bf16*)take((size_t)MID_ * V_ * 2);
    float*  qbuf = (float*)take((size_t)B_ * T_ * KQ_ * 4);
    float*  kbuf = (float*)take((size_t)L_ * B_ * T_ * KQ_ * 4);
    float*  vbuf = (float*)take((size_t)L_ * B_ * T_ * H_ * 4);
    __bf16* aggB = (__bf16*)take((size_t)B_ * T_ * H_ * 2);
    __bf16* midB = (__bf16*)take((size_t)B_ * T_ * MID_ * 2);
    __bf16* midI = (__bf16*)take((size_t)B_ * S_ * MID_ * 2);

    // 1) weights -> bf16, TRANSPOSED to (N x K) so GEMM staging is pure b128
    auto castT = [&](const float* src, __bf16* dst, int K, int N) {
        cast_transpose_kernel<<<dim3(N / 32, K / 32), 256, 0, stream>>>(src, dst, K, N);
    };
    castT(Wq, WqT, D_, KQ_);
    castT(Wk, WkT, D_, KQ_);
    castT(Wv, WvT, D_, H_);
    castT(W1, W1T, H_, MID_);
    castT(W2, W2T, MID_, V_);

    // 2) stride-sample h -> bf16 (+ layer mean in slot L)
    {
        int groups = (B_ * T_ * D_) / 4;
        pack_h_kernel<<<(groups + 255) / 256, 256, 0, stream>>>(hs, hB);
    }

    // 3) q = relu(hmean @ Wq + bq)           (512 x 8192 x 128)
    gemm_bf16_kernel<1, 0><<<dim3((B_ * T_) / 128, KQ_ / 128), 256, 0, stream>>>(
        hB + (size_t)L_ * B_ * T_ * D_, WqT, bq, qbuf, B_ * T_, KQ_, D_);
    // 4) k = relu(h @ Wk + bk)               (3072 x 8192 x 128)
    gemm_bf16_kernel<1, 0><<<dim3((L_ * B_ * T_) / 128, KQ_ / 128), 256, 0, stream>>>(
        hB, WkT, bk, kbuf, L_ * B_ * T_, KQ_, D_);
    // 5) v = h @ Wv + bv                     (3072 x 8192 x 512)
    gemm_bf16_kernel<0, 0><<<dim3((L_ * B_ * T_) / 128, H_ / 128), 256, 0, stream>>>(
        hB, WvT, bv, vbuf, L_ * B_ * T_, H_, D_);
    // 6) softmax over layers + weighted aggregation -> bf16
    attn_kernel<<<B_ * T_, 128, 0, stream>>>(qbuf, kbuf, vbuf, aggB);
    // 7) mid = relu(agg @ W1 + b1) -> bf16   (512 x 512 x 256)
    gemm_bf16_kernel<1, 1><<<dim3((B_ * T_) / 128, MID_ / 128), 256, 0, stream>>>(
        aggB, W1T, b1, midB, B_ * T_, MID_, H_);
    // 8) interpolate mid to length S (linear interp commutes with @W2 + b2)
    interp_kernel<<<B_ * S_, MID_, 0, stream>>>(midB, midI);
    // 9) logits = midI @ W2 + b2 -> d_out    (2048 x 256 x 32000, 262MB write)
    gemm_bf16_kernel<0, 0><<<dim3((B_ * S_) / 128, V_ / 128), 256, 0, stream>>>(
        midI, W2T, b2, out, B_ * S_, V_, MID_);
    // 10) sampled_indices tail of the tuple output
    tail_kernel<<<1, 256, 0, stream>>>(out, out_size);
}
